// LSTMEncoder_39444979647089
// MI455X (gfx1250) — matrix operand order repaired and checked by reading, not compiled, not run on hardware
//
#include <hip/hip_runtime.h>

// Problem constants (match reference)
#define Bn 256
#define Tt 512
#define Ff 256
#define Hh 1024
#define Ll 128
#define Gg 4096   // 4*H
#define KC 1280   // F + H

typedef __attribute__((ext_vector_type(16))) __bf16 v16bf;
typedef __attribute__((ext_vector_type(8)))  float  v8f;

union FragU { uint4 u[2]; v16bf v; };
union Pack8 { __bf16 h[8]; uint4 u; };

#define LDS_STRIDE 40   // 32 bf16 data + 8 pad -> 80B rows, conflict-free b128 ds loads
#define ZS_STRIDE  18   // float stride for gate-exchange tile (bank-friendly)
#define ASZ (128 * LDS_STRIDE)   // A tile elems (128 rows x 32 k)
#define BSZ (64 * LDS_STRIDE)    // B tile elems (64 rows x 32 k)

// A fragment (16x32 bf16): lanes 0-15 hold rows; lower half K0-7 & K16-23,
// upper half K8-15 & K24-31  (ISA 7.12.2)
__device__ inline v16bf load_fragA_lds(const __bf16* base, int row0, int lane) {
  const int r = lane & 15, half = lane >> 4;
  const __bf16* p = base + (row0 + r) * LDS_STRIDE + half * 8;
  FragU f;
  f.u[0] = *(const uint4*)p;
  f.u[1] = *(const uint4*)(p + 16);
  return f.v;
}
// B fragment (32x16 bf16) from N-major tile: lanes hold columns; lower half
// K0-7 & K16-23, upper half K8-15 & K24-31 (matches A fragment)
__device__ inline v16bf load_fragB_lds(const __bf16* base, int row0, int lane) {
  const int r = lane & 15, half = lane >> 4;
  const __bf16* p = base + (row0 + r) * LDS_STRIDE + half * 8;
  FragU f;
  f.u[0] = *(const uint4*)p;
  f.u[1] = *(const uint4*)(p + 16);
  return f.v;
}

__device__ inline float sigmoidf(float x) { return 1.0f / (1.0f + __expf(-x)); }
__device__ inline float tanh_fast(float x) {
  x = fminf(15.0f, fmaxf(-15.0f, x));
  float e = __expf(2.0f * x);
  return (e - 1.0f) / (e + 1.0f);
}

// Two-level device-wide barrier: 16-block groups -> root, counters on separate
// 128B lines; agent-scope atomics; release fence on arrive, acquire on wake.
// base[0]=gen, base[8]=root, base[32 + g*32]=group counter g.
__device__ inline void grid_sync_tree(int* base, int bid, int nGroups) {
  __syncthreads();
  if (threadIdx.x == 0) {
    __threadfence();   // release h writes to agent scope
    int* gen  = base;
    int* root = base + 8;
    int* grp  = base + 32 + (bid >> 4) * 32;
    int g = __hip_atomic_load(gen, __ATOMIC_RELAXED, __HIP_MEMORY_SCOPE_AGENT);
    bool done = false;
    int a = __hip_atomic_fetch_add(grp, 1, __ATOMIC_ACQ_REL, __HIP_MEMORY_SCOPE_AGENT);
    if (a == 15) {     // last of the group
      __hip_atomic_store(grp, 0, __ATOMIC_RELAXED, __HIP_MEMORY_SCOPE_AGENT);
      int ra = __hip_atomic_fetch_add(root, 1, __ATOMIC_ACQ_REL, __HIP_MEMORY_SCOPE_AGENT);
      if (ra == nGroups - 1) {   // last group: open the barrier
        __hip_atomic_store(root, 0, __ATOMIC_RELAXED, __HIP_MEMORY_SCOPE_AGENT);
        __hip_atomic_fetch_add(gen, 1, __ATOMIC_ACQ_REL, __HIP_MEMORY_SCOPE_AGENT);
        done = true;
      }
    }
    if (!done) {
      while (__hip_atomic_load(gen, __ATOMIC_ACQUIRE, __HIP_MEMORY_SCOPE_AGENT) == g) {
        __builtin_amdgcn_s_sleep(1);
      }
    }
  }
  __syncthreads();
}

// ---------------------------------------------------------------------------
// Persistent fused LSTM: all 512 timesteps in one kernel.
// Grid (2, 64) = 128 blocks: blockIdx.x -> 128 batch rows, blockIdx.y -> 16
// hidden cols j0. 512 threads = 16 waves (4 M x 4 N); wave wn computes gate wn
// for cols j0..j0+15, so gates are block-local (no z round trip). c stays in
// registers; h ping-pongs between two bf16 buffers; one tree-sync per step.
// Inner loop: LDS ping-pong + register prefetch -> one barrier per K-tile.
// ---------------------------------------------------------------------------
__global__ __launch_bounds__(512) void lstm_persistent(
    const float* __restrict__ x,      // (B, T, F) fp32
    const __bf16* __restrict__ WcT,   // (4096 x 1280) N-major bf16 [Wx;Wh]^T
    const float* __restrict__ bias,   // (4096)
    __bf16* __restrict__ hb0,         // (B, H) ping  (zeroed; holds final h)
    __bf16* __restrict__ hb1,         // (B, H) pong
    int* __restrict__ syncBase, int nGroups) {
  __shared__ __bf16 ldsA[2 * ASZ];                 // 20.0 KB (ping-pong)
  __shared__ __bf16 ldsB[2 * BSZ];                 // 10.0 KB (ping-pong)
  __shared__ float  zs[4][64][ZS_STRIDE];          // 18.0 KB (half-tile, 2 passes)

  const int tid  = threadIdx.x;
  const int lane = tid & 31;
  const int wave = tid >> 5;                 // 0..15
  const int wm = wave >> 2, wn = wave & 3;   // 4 M x 4 N waves; wn == gate
  const int mBase = blockIdx.x * 128;
  const int j0 = blockIdx.y * 16;
  const int bid = blockIdx.y * 2 + blockIdx.x;   // 0..127

  // cooperative tile-load indices
  const int arow = tid >> 2;                 // 0..127 (one b128/thread for A)
  const int acol = (tid & 3) * 8;
  const int brow = tid >> 2;                 // 0..63 valid when tid < 256
  const int bcol = (tid & 3) * 8;
  const __bf16* bRowPtr =
      WcT + (size_t)((brow >> 4) * Hh + j0 + (brow & 15)) * KC;

  // gate-phase: thread owns column j0+(tid&15), 2 rows per pass
  const int gcol = tid & 15;
  const int gml0 = (tid >> 4) * 2;           // 0..62
  const float bi  = bias[j0 + gcol];
  const float bf_ = bias[Hh + j0 + gcol];
  const float bg_ = bias[2 * Hh + j0 + gcol];
  const float bo  = bias[3 * Hh + j0 + gcol];

  float creg[4] = {0, 0, 0, 0};              // c state, fixed ownership across t

  for (int t = 0; t < Tt; ++t) {
    const __bf16* hprev = (t & 1) ? hb1 : hb0;
    __bf16* hnext = (t & 1) ? hb0 : hb1;
    const float* xt = x + (size_t)t * Ff;

    v8f acc0{}, acc1{};

    // prefetch K-tile 0 (always the fp32 x segment)
    float4 af0, af1;
    uint4  abf = {0, 0, 0, 0};
    uint4  breg = {0, 0, 0, 0};
    {
      const float* p = xt + (size_t)(mBase + arow) * (Tt * Ff) + acol;
      af0 = *(const float4*)p;
      af1 = *(const float4*)(p + 4);
      if (tid < 256) breg = *(const uint4*)(bRowPtr + bcol);
    }

    int cur = 0;
    for (int k = 0; k < KC; k += 32) {
      // ---- commit staged registers into lds[cur] ----
      if (k < Ff) {
        Pack8 pk;
        pk.h[0] = (__bf16)af0.x; pk.h[1] = (__bf16)af0.y;
        pk.h[2] = (__bf16)af0.z; pk.h[3] = (__bf16)af0.w;
        pk.h[4] = (__bf16)af1.x; pk.h[5] = (__bf16)af1.y;
        pk.h[6] = (__bf16)af1.z; pk.h[7] = (__bf16)af1.w;
        *(uint4*)&ldsA[cur * ASZ + arow * LDS_STRIDE + acol] = pk.u;
      } else {
        *(uint4*)&ldsA[cur * ASZ + arow * LDS_STRIDE + acol] = abf;
      }
      if (tid < 256)
        *(uint4*)&ldsB[cur * BSZ + brow * LDS_STRIDE + bcol] = breg;
      // ---- prefetch next K-tile (overlaps with the WMMAs below) ----
      const int kn = k + 32;
      if (kn < KC) {
        if (kn < Ff) {
          const float* p = xt + (size_t)(mBase + arow) * (Tt * Ff) + kn + acol;
          af0 = *(const float4*)p;
          af1 = *(const float4*)(p + 4);
        } else {
          abf = *(const uint4*)(hprev + (size_t)(mBase + arow) * Hh + (kn - Ff) + acol);
        }
        if (tid < 256) breg = *(const uint4*)(bRowPtr + kn + bcol);
      }
      __syncthreads();                       // tile k visible block-wide

      const __bf16* pa = ldsA + cur * ASZ;
      const __bf16* pb = ldsB + cur * BSZ;
      v16bf a0 = load_fragA_lds(pa, wm * 32,      lane);
      v16bf a1 = load_fragA_lds(pa, wm * 32 + 16, lane);
      v16bf b0 = load_fragB_lds(pb, wn * 16,      lane);
      acc0 = __builtin_amdgcn_wmma_f32_16x16x32_bf16(false, a0, false, b0, (short)0, acc0, false, false);
      acc1 = __builtin_amdgcn_wmma_f32_16x16x32_bf16(false, a1, false, b0, (short)0, acc1, false, false);
      cur ^= 1;
    }

    // ---- gates, block-local, two row-halves to bound LDS ----
    const int r15 = lane & 15, half = lane >> 4;
#pragma unroll
    for (int pass = 0; pass < 2; ++pass) {
      __syncthreads();                       // zs free
      if ((wm >> 1) == pass) {               // waves owning rows pass*64..+63
        const int lr = (wm & 1) * 32 + half * 8;
#pragma unroll
        for (int r = 0; r < 8; ++r) {
          zs[wn][lr + r][r15]      = acc0[r];
          zs[wn][lr + 16 + r][r15] = acc1[r];
        }
      }
      __syncthreads();                       // zs ready
#pragma unroll
      for (int r = 0; r < 2; ++r) {
        const int ml = gml0 + r;             // 0..63 within this half
        float zi = zs[0][ml][gcol] + bi;
        float zf = zs[1][ml][gcol] + bf_;
        float zg = zs[2][ml][gcol] + bg_;
        float zo = zs[3][ml][gcol] + bo;
        float ig = sigmoidf(zi);
        float fg = sigmoidf(zf);
        float gg = tanh_fast(zg);
        float og = sigmoidf(zo);
        float cn = fg * creg[pass * 2 + r] + ig * gg;
        creg[pass * 2 + r] = cn;
        const int m = mBase + pass * 64 + ml;
        hnext[(size_t)m * Hh + j0 + gcol] = (__bf16)(og * tanh_fast(cn));
      }
    }

    if (t != Tt - 1) grid_sync_tree(syncBase, bid, nGroups);
  }
}

// ---------------------------------------------------------------------------
// Dense head GEMM (bf16 WMMA), block tile 64x128, 8 waves, wave tile 32x32.
// mode 1: bias+relu -> bf16; mode 2: bias + (mu, log_sigma) split -> fp32 out.
// ---------------------------------------------------------------------------
__device__ inline void store_tile(v8f a, int m0, int n0, int lane,
                                  int N, int M, const float* __restrict__ bias,
                                  float* __restrict__ Cout,
                                  __bf16* __restrict__ CoutB, int mode) {
  const int r15 = lane & 15, half = lane >> 4;
  const int n = n0 + r15;
  const int mrow = m0 + half * 8;
#pragma unroll
  for (int r = 0; r < 8; ++r) {
    float v = a[r] + bias[n];
    const int m = mrow + r;
    if (mode == 1) {
      v = v > 0.0f ? v : 0.0f;
      CoutB[(size_t)m * N + n] = (__bf16)v;
    } else {
      const int hN = N >> 1;
      size_t o = (n < hN) ? ((size_t)m * hN + n)
                          : ((size_t)M * hN + (size_t)m * hN + (n - hN));
      Cout[o] = v;
    }
  }
}

__global__ __launch_bounds__(256) void gemm_wmma(
    const __bf16* __restrict__ A, int lda,
    const __bf16* __restrict__ Bt, int ldb,
    int K, int N, int M,
    const float* __restrict__ bias,
    float* __restrict__ Cout, __bf16* __restrict__ CoutB, int mode) {
  __shared__ __bf16 ldsA[64 * LDS_STRIDE];
  __shared__ __bf16 ldsB[128 * LDS_STRIDE];

  const int tid  = threadIdx.x;
  const int lane = tid & 31;
  const int wave = tid >> 5;
  const int wm = wave >> 2, wn = wave & 3;
  const int mBase = blockIdx.x * 64;
  const int nBase = blockIdx.y * 128;

  const int arow  = tid >> 2;
  const int acol  = (tid & 3) * 8;
  const int brow0 = tid >> 2;
  const int brow1 = (tid + 256) >> 2;
  const int bcol  = (tid & 3) * 8;

  v8f acc00{}, acc01{}, acc10{}, acc11{};

  for (int k = 0; k < K; k += 32) {
    __syncthreads();
    *(uint4*)&ldsA[arow * LDS_STRIDE + acol] =
        *(const uint4*)(A + (size_t)(mBase + arow) * lda + k + acol);
    *(uint4*)&ldsB[brow0 * LDS_STRIDE + bcol] =
        *(const uint4*)(Bt + (size_t)(nBase + brow0) * ldb + k + bcol);
    *(uint4*)&ldsB[brow1 * LDS_STRIDE + bcol] =
        *(const uint4*)(Bt + (size_t)(nBase + brow1) * ldb + k + bcol);
    __syncthreads();

    v16bf a0 = load_fragA_lds(ldsA, wm * 32,      lane);
    v16bf a1 = load_fragA_lds(ldsA, wm * 32 + 16, lane);
    v16bf b0 = load_fragB_lds(ldsB, wn * 32,      lane);
    v16bf b1 = load_fragB_lds(ldsB, wn * 32 + 16, lane);
    acc00 = __builtin_amdgcn_wmma_f32_16x16x32_bf16(false, a0, false, b0, (short)0, acc00, false, false);
    acc01 = __builtin_amdgcn_wmma_f32_16x16x32_bf16(false, a0, false, b1, (short)0, acc01, false, false);
    acc10 = __builtin_amdgcn_wmma_f32_16x16x32_bf16(false, a1, false, b0, (short)0, acc10, false, false);
    acc11 = __builtin_amdgcn_wmma_f32_16x16x32_bf16(false, a1, false, b1, (short)0, acc11, false, false);
  }

  store_tile(acc00, mBase + wm * 32,      nBase + wn * 32,      lane, N, M, bias, Cout, CoutB, mode);
  store_tile(acc01, mBase + wm * 32,      nBase + wn * 32 + 16, lane, N, M, bias, Cout, CoutB, mode);
  store_tile(acc10, mBase + wm * 32 + 16, nBase + wn * 32,      lane, N, M, bias, Cout, CoutB, mode);
  store_tile(acc11, mBase + wm * 32 + 16, nBase + wn * 32 + 16, lane, N, M, bias, Cout, CoutB, mode);
}

// dst[n*ldd + koff + k] = (bf16) src[k*N + n]   (weights -> N-major bf16)
__global__ __launch_bounds__(256) void transpose_cvt(
    const float* __restrict__ src, __bf16* __restrict__ dst,
    int K, int N, int ldd, int koff) {
  long idx = (long)blockIdx.x * 256 + threadIdx.x;
  if (idx >= (long)K * N) return;
  int k = (int)(idx / N), n = (int)(idx % N);
  dst[(size_t)n * ldd + koff + k] = (__bf16)src[idx];
}

extern "C" void kernel_launch(void* const* d_in, const int* in_sizes, int n_in,
                              void* d_out, int out_size, void* d_ws, size_t ws_size,
                              hipStream_t stream) {
  (void)in_sizes; (void)n_in; (void)out_size; (void)ws_size;
  const float* x  = (const float*)d_in[0];
  const float* Wx = (const float*)d_in[1];
  const float* Wh = (const float*)d_in[2];
  const float* b  = (const float*)d_in[3];
  const float* W1 = (const float*)d_in[4];
  const float* b1 = (const float*)d_in[5];
  const float* W2 = (const float*)d_in[6];
  const float* b2 = (const float*)d_in[7];
  float* out = (float*)d_out;

  // workspace carve-out (~15 MB)
  char* ws = (char*)d_ws;
  size_t off = 0;
  auto carve = [&](size_t bytes) -> void* {
    void* p = ws + off;
    off = (off + bytes + 255) & ~(size_t)255;
    return p;
  };
  __bf16* WcombT = (__bf16*)carve((size_t)Gg * KC * 2);   // [Wx;Wh]^T, N-major
  __bf16* W1T    = (__bf16*)carve((size_t)Hh * Hh * 2);
  __bf16* W2T    = (__bf16*)carve((size_t)(2 * Ll) * Hh * 2);
  __bf16* hb0    = (__bf16*)carve((size_t)Bn * Hh * 2);
  __bf16* hb1    = (__bf16*)carve((size_t)Bn * Hh * 2);
  __bf16* out1b  = (__bf16*)carve((size_t)Bn * Hh * 2);
  int*    syncP  = (int*)carve(2048);   // tree-barrier state (gen/root/group cnts)

  // one-time weight transpose+convert
  { long n = (long)Ff * Gg;     transpose_cvt<<<(n + 255) / 256, 256, 0, stream>>>(Wx, WcombT, Ff, Gg, KC, 0);  }
  { long n = (long)Hh * Gg;     transpose_cvt<<<(n + 255) / 256, 256, 0, stream>>>(Wh, WcombT, Hh, Gg, KC, Ff); }
  { long n = (long)Hh * Hh;     transpose_cvt<<<(n + 255) / 256, 256, 0, stream>>>(W1, W1T, Hh, Hh, Hh, 0);     }
  { long n = (long)Hh * 2 * Ll; transpose_cvt<<<(n + 255) / 256, 256, 0, stream>>>(W2, W2T, Hh, 2 * Ll, Hh, 0); }
  hipMemsetAsync(hb0, 0, (size_t)Bn * Hh * 2, stream);   // h0 = 0 (c0 = 0 in regs)
  hipMemsetAsync(syncP, 0, 2048, stream);                // barrier state

  // the whole recurrence: one persistent kernel, 128 blocks x 512 threads
  const int nGroups = ((Bn / 128) * (Hh / 16)) / 16;     // 128 blocks / 16 = 8
  lstm_persistent<<<dim3(Bn / 128, Hh / 16), 512, 0, stream>>>(
      x, WcombT, b, hb0, hb1, syncP, nGroups);
  // T=512 even: final h lands in hb0

  // out1 = relu(h @ W1 + b1) -> bf16
  gemm_wmma<<<dim3(Bn / 64, Hh / 128), 256, 0, stream>>>(
      hb0, Hh, W1T, Hh, Hh, Hh, Bn, b1, nullptr, out1b, /*mode=*/1);

  // out2 = out1 @ W2 + b2 -> (mu, log_sigma) split into d_out
  gemm_wmma<<<dim3(Bn / 64, (2 * Ll) / 128), 256, 0, stream>>>(
      out1b, Hh, W2T, Hh, Hh, 2 * Ll, Bn, b2, out, nullptr, /*mode=*/2);
}
